// SpatialReductionAttention_31447750541387
// MI455X (gfx1250) — matrix operand order repaired and checked by reading, not compiled, not run on hardware
//
#include <hip/hip_runtime.h>

// ---------------------------------------------------------------------------
// Spatial-Reduction Attention for MI455X (gfx1250, wave32, WMMA bf16)
// Pipeline: cast->bf16 | q=1x1conv GEMM | k,v=4x4/s4 implicit GEMM |
//           fused attention (QK^T, softmax, PV) | out=1x1conv GEMM (f32 out)
// All GEMMs use v_wmma_f32_16x16x32_bf16 (K=32 == HEAD_DIM for the S-GEMM).
// v2: 16x64 wave tiles (4 WMMA per A-frag), single-barrier LDS double
//     buffering, b64 im2col moves, global_prefetch_b8 for streamed operand.
// ---------------------------------------------------------------------------

typedef __bf16 bf16;
typedef __attribute__((ext_vector_type(16))) __bf16 v16bf;
typedef __attribute__((ext_vector_type(8)))  __bf16 v8bf;
typedef __attribute__((ext_vector_type(4)))  __bf16 v4bf;
typedef __attribute__((ext_vector_type(8)))  float  v8f;

union V16U { v16bf v; struct { v8bf lo; v8bf hi; } h; };
union V8U  { v8bf v; unsigned short u[8]; };

#define WMMA_BF16(A, B, C) \
  __builtin_amdgcn_wmma_f32_16x16x32_bf16(false, (A), false, (B), (short)0, (C), false, false)

// ---------------------------------------------------------------- cast f32->bf16
__global__ void cast_f32_bf16(const float* __restrict__ s, bf16* __restrict__ d, int n) {
  int i = blockIdx.x * blockDim.x + threadIdx.x;
  int stride = gridDim.x * blockDim.x;
  for (; i < n; i += stride) d[i] = (bf16)s[i];
}

// ---------------------------------------------------------------- 1x1 conv GEMM
// Y[b] = W(256x256) * X[b](256x4096). Wave tile = 16M x 64N (4 acc, 1 A-frag).
// LDS double buffer -> one barrier per k-step. blocks = B*64*2 = 1024.
template <bool F32OUT>
__global__ __launch_bounds__(256)
void gemm1x1(const bf16* __restrict__ W, const bf16* __restrict__ X,
             bf16* __restrict__ Yb, float* __restrict__ Yf) {
  __shared__ bf16 XT[2][64 * 32];         // [n][k] transposed tile, x2 buffers
  const int t  = threadIdx.x;
  const int w  = t >> 5, l = t & 31;
  const int bid = blockIdx.x;
  const int nbk = bid & 63;
  const int mb  = (bid >> 6) & 1;
  const int b   = bid >> 7;
  const int n0  = nbk * 64;
  const int m0  = (mb * 8 + w) * 16;
  const int lN = l & 15, lH = l >> 4;
  const bf16* Xb = X + (size_t)b * 256 * 4096;
  const int kl = t >> 3;                  // coop: k_local 0..31
  const int nb8 = (8 * t) & 63;           // coop: 8-wide n run
  v8f acc0 = {}, acc1 = {}, acc2 = {}, acc3 = {};

  // prologue: stage tile 0
  {
    V8U r; r.v = *(const v8bf*)(Xb + (size_t)kl * 4096 + n0 + nb8);
    unsigned short* s = (unsigned short*)XT[0];
#pragma unroll
    for (int i = 0; i < 8; ++i) s[(nb8 + i) * 32 + kl] = r.u[i];
  }
  __syncthreads();

#pragma unroll
  for (int s8 = 0; s8 < 8; ++s8) {
    const int kk = s8 * 32;
    // stage next tile into the other buffer (no barrier needed before compute)
    if (s8 < 7) {
      V8U r; r.v = *(const v8bf*)(Xb + (size_t)(kk + 32 + kl) * 4096 + n0 + nb8);
      unsigned short* s = (unsigned short*)XT[(s8 + 1) & 1];
#pragma unroll
      for (int i = 0; i < 8; ++i) s[(nb8 + i) * 32 + kl] = r.u[i];
    }
    if (s8 < 6)
      __builtin_prefetch(Xb + (size_t)(kk + 64 + kl) * 4096 + n0 + nb8);
    // compute from current buffer
    const bf16* buf = XT[s8 & 1];
    V16U a;
    const bf16* ap = W + (size_t)(m0 + lN) * 256 + kk + lH * 8;
    a.h.lo = *(const v8bf*)ap;            // K = half*8 .. +7
    a.h.hi = *(const v8bf*)(ap + 16);     // K = 16+half*8 .. +7
    V16U b0, b1, b2, b3;
    b0.h.lo = *(const v8bf*)&buf[( 0 + lN) * 32 + lH * 8];
    b0.h.hi = *(const v8bf*)&buf[( 0 + lN) * 32 + 16 + lH * 8];
    b1.h.lo = *(const v8bf*)&buf[(16 + lN) * 32 + lH * 8];
    b1.h.hi = *(const v8bf*)&buf[(16 + lN) * 32 + 16 + lH * 8];
    b2.h.lo = *(const v8bf*)&buf[(32 + lN) * 32 + lH * 8];
    b2.h.hi = *(const v8bf*)&buf[(32 + lN) * 32 + 16 + lH * 8];
    b3.h.lo = *(const v8bf*)&buf[(48 + lN) * 32 + lH * 8];
    b3.h.hi = *(const v8bf*)&buf[(48 + lN) * 32 + 16 + lH * 8];
    acc0 = WMMA_BF16(a.v, b0.v, acc0);
    acc1 = WMMA_BF16(a.v, b1.v, acc1);
    acc2 = WMMA_BF16(a.v, b2.v, acc2);
    acc3 = WMMA_BF16(a.v, b3.v, acc3);
    if (s8 < 7) __syncthreads();
  }

  const size_t ob = ((size_t)b * 256 + m0) * 4096 + n0;
#pragma unroll
  for (int r = 0; r < 8; ++r) {
    const size_t row = ob + (size_t)(r + 8 * lH) * 4096;
    if (F32OUT) {
      Yf[row +  0 + lN] = acc0[r];
      Yf[row + 16 + lN] = acc1[r];
      Yf[row + 32 + lN] = acc2[r];
      Yf[row + 48 + lN] = acc3[r];
    } else {
      Yb[row +  0 + lN] = (bf16)acc0[r];
      Yb[row + 16 + lN] = (bf16)acc1[r];
      Yb[row + 32 + lN] = (bf16)acc2[r];
      Yb[row + 48 + lN] = (bf16)acc3[r];
    }
  }
}

// ---------------------------------------------------------------- 4x4/s4 conv
// Implicit GEMM: K = c*16+kh*4+kw (wk OIHW flattening). Wave tile = 16M x 32N.
// kw-groups of 4 are contiguous in memory -> b64 moves. blocks = B*8*2 = 128.
__global__ __launch_bounds__(256)
void conv4x4(const bf16* __restrict__ W, const bf16* __restrict__ X,
             bf16* __restrict__ Y) {
  __shared__ bf16 XT[2][32 * 32];         // [j][k] tile, x2 buffers
  const int t = threadIdx.x, w = t >> 5, l = t & 31;
  const int bid = blockIdx.x;
  const int jb = bid & 7, mb = (bid >> 3) & 1, b = bid >> 4;
  const int j0 = jb * 32, m0 = (mb * 8 + w) * 16;
  const int lN = l & 15, lH = l >> 4;
  const bf16* Xb = X + (size_t)b * 256 * 4096;
  // coop: one kw-group (4 contiguous bf16) per thread per step
  const int jl  = t >> 3;                 // j_local 0..31
  const int kg4 = (t & 7) * 4;            // k_local group base
  const int hp = (j0 + jl) >> 4, wp = (j0 + jl) & 15;
  const bf16* gsrc0 = Xb + (4 * hp) * 64 + 4 * wp;   // + c*4096 + kh*64 later
  v8f acc0 = {}, acc1 = {};

  auto stage = [&](int kk, int buf) {
    int kg = kk + kg4;
    int c = kg >> 4, kh = (kg >> 2) & 3;
    v4bf r = *(const v4bf*)(gsrc0 + (size_t)c * 4096 + kh * 64);
    *(v4bf*)&XT[buf][jl * 32 + kg4] = r;
  };

  stage(0, 0);
  __syncthreads();

  for (int s8 = 0; s8 < 128; ++s8) {
    const int kk = s8 * 32;
    if (s8 < 127) stage(kk + 32, (s8 + 1) & 1);
    if (s8 < 126) {
      int kg = kk + 64 + kg4;
      __builtin_prefetch(gsrc0 + (size_t)(kg >> 4) * 4096 + ((kg >> 2) & 3) * 64);
    }
    const bf16* buf = XT[s8 & 1];
    V16U a;
    const bf16* ap = W + (size_t)(m0 + lN) * 4096 + kk + lH * 8;
    a.h.lo = *(const v8bf*)ap;
    a.h.hi = *(const v8bf*)(ap + 16);
    V16U b0, b1;
    b0.h.lo = *(const v8bf*)&buf[( 0 + lN) * 32 + lH * 8];
    b0.h.hi = *(const v8bf*)&buf[( 0 + lN) * 32 + 16 + lH * 8];
    b1.h.lo = *(const v8bf*)&buf[(16 + lN) * 32 + lH * 8];
    b1.h.hi = *(const v8bf*)&buf[(16 + lN) * 32 + 16 + lH * 8];
    acc0 = WMMA_BF16(a.v, b0.v, acc0);
    acc1 = WMMA_BF16(a.v, b1.v, acc1);
    if (s8 < 127) __syncthreads();
  }

  const size_t ob = ((size_t)b * 256 + m0) * 256 + j0;
#pragma unroll
  for (int r = 0; r < 8; ++r) {
    const size_t row = ob + (size_t)(r + 8 * lH) * 256;
    Y[row +  0 + lN] = (bf16)acc0[r];
    Y[row + 16 + lN] = (bf16)acc1[r];
  }
}

// ---------------------------------------------------------------- attention
// One block = one (b, head, 128-row i-block). 8 waves, each owns 16 i-rows.
// S = q^T k (K=32, one WMMA per 16x16 tile), softmax over 256 j, O = P v^T.
__global__ __launch_bounds__(256)
void sra_attention(const bf16* __restrict__ Q, const bf16* __restrict__ K,
                   const bf16* __restrict__ V, bf16* __restrict__ O) {
  __shared__ bf16 kT[256 * 32];           // [j][d]  (B-frag reads contiguous)
  __shared__ bf16 vL[32 * 256];           // [d][j]  (natural layout)
  __shared__ bf16 qT[128 * 32];           // [i_local][d] (A-frag reads contiguous)
  __shared__ bf16 Pch[8][16 * 32];        // per-wave P re-layout chunk
  const int t = threadIdx.x, w = t >> 5, l = t & 31;
  const int bid = blockIdx.x;
  const int ib = bid & 31, h = (bid >> 5) & 7, b = bid >> 8;
  const int i0 = ib * 128;
  const int lN = l & 15, lH = l >> 4;
  const float SCALE = 0.17677669529663687f;  // 1/sqrt(32)
  const size_t ch0 = (size_t)b * 256 + h * 32;

  // cooperative staging --------------------------------------------------
  {                                        // vL: straight 16KB copy
    const v8bf* s = (const v8bf*)(V + ch0 * 256);
    v8bf* d = (v8bf*)vL;
#pragma unroll
    for (int r = 0; r < 4; ++r) d[t * 4 + r] = s[t * 4 + r];
  }
  {                                        // kT: transpose 32x256 -> [j][d]
    const int d0 = t >> 3, jb = (t & 7) * 32;
    const bf16* s = K + (ch0 + d0) * 256 + jb;
#pragma unroll
    for (int c = 0; c < 32; ++c) kT[(jb + c) * 32 + d0] = s[c];
  }
  {                                        // qT: transpose 32x128 -> [i][d]
    const int d0 = t >> 3, is = (t & 7) * 16;
    const bf16* s = Q + (ch0 + d0) * 4096 + i0 + is;
#pragma unroll
    for (int c = 0; c < 16; ++c) qT[(is + c) * 32 + d0] = s[c];
  }
  __syncthreads();

  // S = q^T k  ------------------------------------------------------------
  V16U a;
  {
    const bf16* ap = &qT[(w * 16 + lN) * 32 + lH * 8];
    a.h.lo = *(const v8bf*)ap;
    a.h.hi = *(const v8bf*)(ap + 16);
  }
  v8f S[16];
#pragma unroll
  for (int jt = 0; jt < 16; ++jt) {
    V16U bm;
    bm.h.lo = *(const v8bf*)&kT[(jt * 16 + lN) * 32 + lH * 8];
    bm.h.hi = *(const v8bf*)&kT[(jt * 16 + lN) * 32 + 16 + lH * 8];
    v8f z = {};
    S[jt] = WMMA_BF16(a.v, bm.v, z);
  }

  // softmax (row = r + 8*lH lives across the 16 lanes sharing lH) ---------
#pragma unroll
  for (int jt = 0; jt < 16; ++jt)
#pragma unroll
    for (int r = 0; r < 8; ++r) S[jt][r] *= SCALE;
  float inv[8];
#pragma unroll
  for (int r = 0; r < 8; ++r) {
    float m = -3.0e38f;
#pragma unroll
    for (int jt = 0; jt < 16; ++jt) m = fmaxf(m, S[jt][r]);
    for (int s = 1; s < 16; s <<= 1) m = fmaxf(m, __shfl_xor(m, s, 32));
    float sum = 0.f;
#pragma unroll
    for (int jt = 0; jt < 16; ++jt) {
      float e = __expf(S[jt][r] - m);
      S[jt][r] = e;
      sum += e;
    }
    for (int s = 1; s < 16; s <<= 1) sum += __shfl_xor(sum, s, 32);
    inv[r] = 1.0f / sum;
  }

  // O = P * v^T (K=256 in 8 steps; per-wave LDS chunk re-layouts P) -------
  v8f o0 = {}, o1 = {};
  bf16* P = Pch[w];
  for (int jj = 0; jj < 8; ++jj) {
#pragma unroll
    for (int jt2 = 0; jt2 < 2; ++jt2) {
      int jt = jj * 2 + jt2;
#pragma unroll
      for (int r = 0; r < 8; ++r)
        P[(r + 8 * lH) * 32 + jt2 * 16 + lN] = (bf16)(S[jt][r] * inv[r]);
    }
    V16U pa, b0, b1;
    const bf16* pp = &P[lN * 32 + lH * 8];   // same-wave DS ops are in-order
    pa.h.lo = *(const v8bf*)pp;
    pa.h.hi = *(const v8bf*)(pp + 16);
    b0.h.lo = *(const v8bf*)&vL[(     lN) * 256 + jj * 32 + lH * 8];
    b0.h.hi = *(const v8bf*)&vL[(     lN) * 256 + jj * 32 + 16 + lH * 8];
    b1.h.lo = *(const v8bf*)&vL[(16 + lN) * 256 + jj * 32 + lH * 8];
    b1.h.hi = *(const v8bf*)&vL[(16 + lN) * 256 + jj * 32 + 16 + lH * 8];
    o0 = WMMA_BF16(pa.v, b0.v, o0);
    o1 = WMMA_BF16(pa.v, b1.v, o1);
  }

  // store O as [B, 256ch, 4096] bf16 --------------------------------------
  const size_t ob = ch0 * 4096 + i0 + w * 16;
#pragma unroll
  for (int r = 0; r < 8; ++r) {
    int irow = r + 8 * lH;
    O[ob + (size_t)(     lN) * 4096 + irow] = (bf16)o0[r];
    O[ob + (size_t)(16 + lN) * 4096 + irow] = (bf16)o1[r];
  }
}

// ---------------------------------------------------------------- launcher
extern "C" void kernel_launch(void* const* d_in, const int* in_sizes, int n_in,
                              void* d_out, int out_size, void* d_ws, size_t ws_size,
                              hipStream_t stream) {
  const float* x    = (const float*)d_in[0];
  const float* wq   = (const float*)d_in[1];
  const float* wk   = (const float*)d_in[2];
  const float* wv   = (const float*)d_in[3];
  const float* wout = (const float*)d_in[4];
  float* out = (float*)d_out;

  char* ws = (char*)d_ws;
  size_t off = 0;
  auto alloc = [&](size_t elems) {
    bf16* p = (bf16*)(ws + off);
    off += ((elems * sizeof(bf16) + 255) / 256) * 256;
    return p;
  };
  bf16* xb    = alloc(8ull * 256 * 4096);   // x bf16
  bf16* wqb   = alloc(256 * 256);
  bf16* wkb   = alloc(256ull * 256 * 16);
  bf16* wvb   = alloc(256ull * 256 * 16);
  bf16* woutb = alloc(256 * 256);
  bf16* qb    = alloc(8ull * 256 * 4096);   // q
  bf16* kb    = alloc(8ull * 256 * 256);    // k (spatial-reduced)
  bf16* vb    = alloc(8ull * 256 * 256);    // v
  bf16* obuf  = alloc(8ull * 256 * 4096);   // attention output
  (void)ws_size;

  cast_f32_bf16<<<2048, 256, 0, stream>>>(x,    xb,    8 * 256 * 4096);
  cast_f32_bf16<<<64,   256, 0, stream>>>(wq,   wqb,   256 * 256);
  cast_f32_bf16<<<512,  256, 0, stream>>>(wk,   wkb,   256 * 256 * 16);
  cast_f32_bf16<<<512,  256, 0, stream>>>(wv,   wvb,   256 * 256 * 16);
  cast_f32_bf16<<<64,   256, 0, stream>>>(wout, woutb, 256 * 256);

  gemm1x1<false><<<1024, 256, 0, stream>>>(wqb, xb, qb, nullptr);
  conv4x4<<<128, 256, 0, stream>>>(wkb, xb, kb);
  conv4x4<<<128, 256, 0, stream>>>(wvb, xb, vb);
  sra_attention<<<2048, 256, 0, stream>>>(qb, kb, vb, obuf);
  gemm1x1<true><<<1024, 256, 0, stream>>>(woutb, obuf, nullptr, out);
}
